// RegPool_85280870629775
// MI455X (gfx1250) — matrix operand order repaired and checked by reading, not compiled, not run on hardware
//
#include <hip/hip_runtime.h>

// Problem constants (match the reference).
#define B_  64
#define R_  50
#define L_  15
#define D_  1024
#define F_  4096
#define M_  (B_ * R_)     // 3200 rows for the GEMM
#define KB_ 32            // K-slab per LDS stage (one WMMA K)
#define LDP 40            // padded LDS row length in bf16 (80B = 20 DWORDs, conflict-free)

typedef __attribute__((ext_vector_type(16))) __bf16 v16bf;
typedef __attribute__((ext_vector_type(8)))  __bf16 v8bf;
typedef __attribute__((ext_vector_type(8)))  float  v8f;

// ---------------------------------------------------------------------------
// Kernel 1: segment-mean over phrase tokens -> out[:, :, D:2D]
// Pure HBM streaming (196.6 MB in): float4 loads, one block per (b,r).
// ---------------------------------------------------------------------------
__global__ __launch_bounds__(256) void phrase_mean_kernel(
    const float* __restrict__ phrases,   // [B,R,L,D]
    const int*   __restrict__ lens,      // [B,R]
    float*       __restrict__ out)       // [B,R,2D]
{
  const int br = blockIdx.x;
  const int t  = threadIdx.x;            // D/4 == 256 float4 slots

  const float4* p = (const float4*)(phrases + (size_t)br * (L_ * D_));
  float4 acc = make_float4(0.f, 0.f, 0.f, 0.f);
#pragma unroll
  for (int l = 0; l < L_; ++l) {
    float4 v = p[l * (D_ / 4) + t];
    acc.x += v.x; acc.y += v.y; acc.z += v.z; acc.w += v.w;
  }
  const float inv = 1.0f / (float)lens[br];
  float4 r = make_float4(acc.x * inv, acc.y * inv, acc.z * inv, acc.w * inv);
  ((float4*)(out + (size_t)br * (2 * D_) + D_))[t] = r;
}

// ---------------------------------------------------------------------------
// Kernel 2: vis = relu(features @ W^T + b) -> out[:, :, 0:D]
// LDS-staged, double-buffered NT GEMM in bf16 (f32 accumulate).
// Block tile 128x128, 4 waves (2x2), 64x64 per wave, K slabs of 32.
// f32 -> bf16 conversion happens ONCE per element at LDS-store time.
// ---------------------------------------------------------------------------
__device__ __forceinline__ void load_tiles(float* __restrict__ ra,
                                           float* __restrict__ rb,
                                           const float* __restrict__ gA,
                                           const float* __restrict__ gB,
                                           int k0)
{
#pragma unroll
  for (int j = 0; j < KB_; ++j) ra[j] = gA[k0 + j];   // -> global_load_b128 x8
#pragma unroll
  for (int j = 0; j < KB_; ++j) rb[j] = gB[k0 + j];   // -> global_load_b128 x8
}

__device__ __forceinline__ void store_tiles(__bf16* __restrict__ sa,
                                            __bf16* __restrict__ sb,
                                            const float* __restrict__ ra,
                                            const float* __restrict__ rb,
                                            int t)
{
#pragma unroll
  for (int c = 0; c < 4; ++c) {
    v8bf va, vb;
#pragma unroll
    for (int j = 0; j < 8; ++j) {
      va[j] = (__bf16)ra[8 * c + j];                  // v_cvt_pk_bf16_f32
      vb[j] = (__bf16)rb[8 * c + j];
    }
    *(v8bf*)(sa + t * LDP + 8 * c) = va;              // ds_store_b128
    *(v8bf*)(sb + t * LDP + 8 * c) = vb;
  }
}

__global__ __launch_bounds__(128) void vis_gemm_kernel(
    const float* __restrict__ feat,   // [M,F]
    const float* __restrict__ Wm,     // [D,F]
    const float* __restrict__ bias,   // [D]
    float*       __restrict__ out)    // [M,2D], cols 0..D-1
{
  // Double-buffered bf16 tiles: 2 * 128*40 * 2B * 2 arrays = 40 KB LDS.
  __shared__ __bf16 sA[2][128 * LDP];
  __shared__ __bf16 sB[2][128 * LDP];

  const int t    = threadIdx.x;
  const int lane = t & 31;
  const int wave = t >> 5;                  // 0..3 (wave32)
  const int wy = wave >> 1, wx = wave & 1;  // 2x2 waves
  const int bm = blockIdx.x * 128;
  const int bn = blockIdx.y * 128;
  const int hh = lane >> 4;                 // lane half
  const int ll = lane & 15;

  // This thread's producer rows (one A row, one W row per K-slab).
  const float* gA = feat + (size_t)(bm + t) * F_;
  const float* gB = Wm   + (size_t)(bn + t) * F_;

  v8f c[4][4] = {};                         // 64x64 wave tile accumulators

  float ra[KB_], rb[KB_];
  load_tiles(ra, rb, gA, gB, 0);            // prologue stage -> buffer 0
  store_tiles(sA[0], sB[0], ra, rb, t);
  __syncthreads();

  int buf = 0;
  for (int k0 = 0; k0 < F_; k0 += KB_) {
    const int nk = k0 + KB_;
    if (nk < F_) load_tiles(ra, rb, gA, gB, nk);  // next slab in flight

    // ---- consume current buffer: ISA §7.12.2 bf16 fragment layouts ----
    const __bf16* sAc = sA[buf];
    const __bf16* sBc = sB[buf];
    v16bf a[4], b[4];
#pragma unroll
    for (int i = 0; i < 4; ++i) {
      const __bf16* base = sAc + (wy * 64 + 16 * i + ll) * LDP;
      v8bf c0 = *(const v8bf*)(base + 8 * hh);        // K = 8h .. 8h+7
      v8bf c1 = *(const v8bf*)(base + 16 + 8 * hh);   // K = 16+8h .. +7
      a[i] = __builtin_shufflevector(c0, c1,
               0,1,2,3,4,5,6,7,8,9,10,11,12,13,14,15);
    }
#pragma unroll
    for (int j = 0; j < 4; ++j) {
      const __bf16* base = sBc + (wx * 64 + 16 * j + ll) * LDP;
      v8bf d0 = *(const v8bf*)(base + 8 * hh);        // K = 8h .. 8h+7
      v8bf d1 = *(const v8bf*)(base + 16 + 8 * hh);   // K = 16+8h .. +7
      b[j] = __builtin_shufflevector(d0, d1,
               0,1,2,3,4,5,6,7,8,9,10,11,12,13,14,15);
    }
#pragma unroll
    for (int i = 0; i < 4; ++i)
#pragma unroll
      for (int j = 0; j < 4; ++j)
        c[i][j] = __builtin_amdgcn_wmma_f32_16x16x32_bf16(
            false, a[i], false, b[j], (short)0, c[i][j], false, false);

    // ---- publish next buffer (waits on global loads, converts once) ----
    if (nk < F_) store_tiles(sA[buf ^ 1], sB[buf ^ 1], ra, rb, t);
    __syncthreads();
    buf ^= 1;
  }

  // Epilogue: bias + ReLU into the vis half of the concat output.
  const int m0w = bm + wy * 64;
  const int n0w = bn + wx * 64;
  const int mh  = 8 * hh;
#pragma unroll
  for (int j = 0; j < 4; ++j) {
    const int n  = n0w + 16 * j + ll;
    const float bv = bias[n];
#pragma unroll
    for (int i = 0; i < 4; ++i) {
      const int mbase = m0w + 16 * i + mh;
#pragma unroll
      for (int r = 0; r < 8; ++r)
        out[(size_t)(mbase + r) * (2 * D_) + n] = fmaxf(c[i][j][r] + bv, 0.0f);
    }
  }
}

// ---------------------------------------------------------------------------
extern "C" void kernel_launch(void* const* d_in, const int* in_sizes, int n_in,
                              void* d_out, int out_size, void* d_ws, size_t ws_size,
                              hipStream_t stream)
{
  (void)in_sizes; (void)n_in; (void)out_size; (void)d_ws; (void)ws_size;

  const float* feat    = (const float*)d_in[0];  // [B,R,F]
  const float* phrases = (const float*)d_in[1];  // [B,R,L,D]
  const float* Wm      = (const float*)d_in[2];  // [D,F]
  const float* bias    = (const float*)d_in[3];  // [D]
  const int*   lens    = (const int*)  d_in[4];  // [B,R]
  float*       out     = (float*)d_out;          // [B,R,2D]

  // Memory-bound half: phrase means.
  phrase_mean_kernel<<<M_, 256, 0, stream>>>(phrases, lens, out);

  // WMMA half: vis GEMM (M=3200, N=1024, K=4096), all tiles exact.
  dim3 grid(M_ / 128, D_ / 128);   // 25 x 8
  vis_gemm_kernel<<<grid, 128, 0, stream>>>(feat, Wm, bias, out);
}
